// ACSeg_79860621902272
// MI455X (gfx1250) — matrix operand order repaired and checked by reading, not compiled, not run on hardware
//
#include <hip/hip_runtime.h>
#include <hip/hip_bf16.h>
#include <math.h>

// ---------------- problem constants ----------------
#define Bb   16
#define Nn   784
#define Dm   384
#define Hh   6
#define Kq   5
#define Lr   3
#define DFf  1536
#define BN   (Bb * Nn)      // 12544
#define DH   (Dm / Hh)      // 64
#define EPSN 1e-8f
#define NPAD 832            // 784 padded up to a multiple of 64 for gram blocking

// output layout (flat fp32): concepts | W | delta | S | M
#define C_OFF   0LL
#define W_OFF   30720LL
#define DLT_OFF 9865216LL
#define S_OFF   19699712LL
#define M_OFF   19762432LL

typedef _Float16 v16h __attribute__((ext_vector_type(16)));
typedef _Float16 h8   __attribute__((ext_vector_type(8)));
typedef float    v8f  __attribute__((ext_vector_type(8)));

// ============== generic small helpers ==============

__global__ void k_f32_to_f16(const float* __restrict__ src,
                             _Float16* __restrict__ dst, int n) {
  int t = blockIdx.x * blockDim.x + threadIdx.x;
  if (t < n) dst[t] = (_Float16)src[t];
}

__global__ void k_zero_f16(_Float16* __restrict__ p, int n) {
  int t = blockIdx.x * blockDim.x + threadIdx.x;
  if (t < n) p[t] = (_Float16)0.f;
}

// dst[o*din + i] = src[i*dout + o]  (store W^T as f16, row-major out x in)
__global__ void k_transpose_f16(const float* __restrict__ src,
                                _Float16* __restrict__ dst, int din, int dout) {
  int t = blockIdx.x * blockDim.x + threadIdx.x;
  if (t >= din * dout) return;
  int o = t / din, i = t % din;
  dst[t] = (_Float16)src[i * dout + o];
}

__global__ void k_broadcast_protos(const float* __restrict__ protos,
                                   float* __restrict__ conc) {
  int t = blockIdx.x * blockDim.x + threadIdx.x;
  if (t < Bb * Kq * Dm) conc[t] = protos[t % (Kq * Dm)];
}

__global__ void k_copy_f32(const float* __restrict__ src,
                           float* __restrict__ dst, int n) {
  int t = blockIdx.x * blockDim.x + threadIdx.x;
  if (t < n) dst[t] = src[t];
}

__global__ void k_zero_f32(float* __restrict__ p, int n) {
  int t = blockIdx.x * blockDim.x + threadIdx.x;
  if (t < n) p[t] = 0.f;
}

// ============== WMMA 32x64 macro-tile (NT layout) ==============
// A: M x K row-major f16 ; Bt: N x K row-major f16 (i.e. B transposed).
// Per CDNA5 ISA 7.12.2 lane layouts:
//  A frag: lane l<16 row l, K = kb+{0..7}, kb+16+{0..7}; lane>=16 K = +8.
//  B frag: lane l<16 col l, K = kb+{0..7}, kb+16+{0..7}; lane>=16 col l-16, K = +8.
// 2x4 register blocking: 8 accumulators (64 VGPRs) + 6 fragments (48 VGPRs)
// stays under the 256 directly-addressable VGPRs (4x4 spilled to scratch),
// while raising arithmetic intensity to ~21 flop/byte (vs 8 for 1 tile/wave).
template <int MV, int NV>
__device__ __forceinline__ void wmma_block_nt(const _Float16* __restrict__ A, int lda,
                                              const _Float16* __restrict__ Bt, int ldb,
                                              int Kd, int lane, v8f (&acc)[MV][NV]) {
  const int r  = lane & 15;
  const int hi = lane >> 4;
  const _Float16* arow = A  + (size_t)r * lda;
  const _Float16* bcol = Bt + (size_t)r * ldb;
  for (int kb = 0; kb < Kd; kb += 32) {
    __builtin_prefetch(arow + kb + 64, 0, 0);   // global_prefetch_b8
    __builtin_prefetch(bcol + kb + 64, 0, 0);
    v16h af[MV], bf[NV];
#pragma unroll
    for (int i = 0; i < MV; ++i) {
      const _Float16* ar = arow + (size_t)(i * 16) * lda + kb;
      h8 lo = *reinterpret_cast<const h8*>(ar + hi * 8);
      h8 hh = *reinterpret_cast<const h8*>(ar + 16 + hi * 8);
#pragma unroll
      for (int e = 0; e < 8; ++e) { af[i][e] = lo[e]; af[i][8 + e] = hh[e]; }
    }
#pragma unroll
    for (int j = 0; j < NV; ++j) {
      const _Float16* br = bcol + (size_t)(j * 16) * ldb + kb;
      h8 blo = *reinterpret_cast<const h8*>(br + hi * 8);
      h8 bhh = *reinterpret_cast<const h8*>(br + 16 + hi * 8);
#pragma unroll
      for (int e = 0; e < 8; ++e) { bf[j][e] = blo[e]; bf[j][8 + e] = bhh[e]; }
    }
#pragma unroll
    for (int i = 0; i < MV; ++i)
#pragma unroll
      for (int j = 0; j < NV; ++j)
        acc[i][j] = __builtin_amdgcn_wmma_f32_16x16x32_f16(false, af[i], false, bf[j],
                                                           (short)0, acc[i][j], false, false);
  }
}

// C[M,N] = A[M,K] @ Bt[N,K]^T + bias ; one wave per 32x64 macro-tile
__global__ void k_wmma_gemm_nt24(const _Float16* __restrict__ A,
                                 const _Float16* __restrict__ Bt,
                                 const float* __restrict__ bias,
                                 float* __restrict__ C,
                                 int M, int Nc, int Kd) {
  int lane = threadIdx.x & 31;
  int wid  = blockIdx.x * (blockDim.x >> 5) + (threadIdx.x >> 5);
  int nst   = Nc >> 6;                 // supertiles along N (64 wide)
  int total = (M >> 5) * nst;          // supertiles along M (32 tall)
  if (wid >= total) return;
  int sm = wid / nst, sn = wid % nst;

  const v8f z = {0.f, 0.f, 0.f, 0.f, 0.f, 0.f, 0.f, 0.f};
  v8f acc[2][4];
#pragma unroll
  for (int i = 0; i < 2; ++i)
#pragma unroll
    for (int j = 0; j < 4; ++j) acc[i][j] = z;

  wmma_block_nt<2, 4>(A + (size_t)(sm * 32) * Kd, Kd,
                      Bt + (size_t)(sn * 64) * Kd, Kd, Kd, lane, acc);

  int r  = lane & 15;
  int hi = lane >> 4;
#pragma unroll
  for (int j = 0; j < 4; ++j) {
    int col = sn * 64 + j * 16 + r;
    float bv = bias ? bias[col] : 0.f;
#pragma unroll
    for (int i = 0; i < 2; ++i) {
#pragma unroll
      for (int p = 0; p < 8; ++p) {
        int row = sm * 32 + i * 16 + p + hi * 8;
        C[(size_t)row * Nc + col] = acc[i][j][p] + bv;
      }
    }
  }
}

// Gram per batch on zero-padded xnh (NPAD rows): A = clip(xn xn^T, 0),
// fused row sums -> Ksum (atomic). Pad rows are zero -> contribute nothing.
__global__ void k_wmma_gram24(const _Float16* __restrict__ xnh,
                              float* __restrict__ Aout,
                              float* __restrict__ Ksum) {
  const int lane = threadIdx.x & 31;
  int wid = blockIdx.x * (blockDim.x >> 5) + (threadIdx.x >> 5);
  const int mst = NPAD / 32;           // 26 supertiles (rows)
  const int nst = NPAD / 64;           // 13 supertiles (cols)
  if (wid >= mst * nst) return;
  int b  = blockIdx.y;
  int si = wid / nst, sj = wid % nst;
  const _Float16* Xb = xnh + (size_t)b * NPAD * Dm;

  const v8f z = {0.f, 0.f, 0.f, 0.f, 0.f, 0.f, 0.f, 0.f};
  v8f acc[2][4];
#pragma unroll
  for (int i = 0; i < 2; ++i)
#pragma unroll
    for (int j = 0; j < 4; ++j) acc[i][j] = z;

  wmma_block_nt<2, 4>(Xb + (size_t)(si * 32) * Dm, Dm,
                      Xb + (size_t)(sj * 64) * Dm, Dm, Dm, lane, acc);

  int r  = lane & 15;
  int hi = lane >> 4;
  float* Ab = Aout + (size_t)b * Nn * Nn;
#pragma unroll
  for (int i = 0; i < 2; ++i) {
#pragma unroll
    for (int p = 0; p < 8; ++p) {
      int row = si * 32 + i * 16 + p + hi * 8;
      float rsum = 0.f;
#pragma unroll
      for (int j = 0; j < 4; ++j) {
        int col = sj * 64 + j * 16 + r;
        float v = fmaxf(acc[i][j][p], 0.f);
        if (row < Nn && col < Nn) Ab[(size_t)row * Nn + col] = v;
        rsum += v;                      // pad cols contribute exactly 0
      }
      // sum across the 16 lanes of this half-wave (A symmetric)
      rsum += __shfl_xor(rsum, 1, 32);
      rsum += __shfl_xor(rsum, 2, 32);
      rsum += __shfl_xor(rsum, 4, 32);
      rsum += __shfl_xor(rsum, 8, 32);
      if (r == 0 && row < Nn) atomicAdd(&Ksum[b * Nn + row], rsum);
    }
  }
}

// ============== small fp32 kernels (concept-path, 80 rows) ==============

__device__ __forceinline__ float gelu_tanh(float x) {
  const float c0 = 0.7978845608028654f;
  float t = c0 * (x + 0.044715f * x * x * x);
  return 0.5f * x * (1.f + tanhf(t));
}

__global__ void k_small_gemm(const float* __restrict__ A,
                             const float* __restrict__ W,
                             const float* __restrict__ bias,
                             const float* __restrict__ addend,
                             float* __restrict__ out,
                             int M, int Din, int Dout, int do_gelu) {
  int t = blockIdx.x * blockDim.x + threadIdx.x;
  if (t >= M * Dout) return;
  int m = t / Dout, o = t % Dout;
  float acc = bias[o];
  const float* a = A + (size_t)m * Din;
  for (int i = 0; i < Din; ++i) acc += a[i] * W[(size_t)i * Dout + o];
  if (do_gelu) acc = gelu_tanh(acc);
  if (addend) acc += addend[t];
  out[t] = acc;
}

__global__ void k_ln_resid(const float* __restrict__ inp,
                           const float* __restrict__ g,
                           const float* __restrict__ bta,
                           const float* __restrict__ resid,
                           float* __restrict__ out) {
  __shared__ float red[256];
  int row = blockIdx.x, tid = threadIdx.x;
  const float* x = inp + (size_t)row * Dm;
  float s = 0.f, s2 = 0.f;
  for (int i = tid; i < Dm; i += 256) { float v = x[i]; s += v; s2 += v * v; }
  red[tid] = s; __syncthreads();
  for (int k = 128; k > 0; k >>= 1) { if (tid < k) red[tid] += red[tid + k]; __syncthreads(); }
  float mean = red[0] / Dm; __syncthreads();
  red[tid] = s2; __syncthreads();
  for (int k = 128; k > 0; k >>= 1) { if (tid < k) red[tid] += red[tid + k]; __syncthreads(); }
  float var = red[0] / Dm - mean * mean;
  float inv = rsqrtf(var + 1e-5f);
  for (int i = tid; i < Dm; i += 256)
    out[(size_t)row * Dm + i] = (x[i] - mean) * inv * g[i] + bta[i] + resid[(size_t)row * Dm + i];
}

// normalize x rows; fp32 flat + f16 mirror into padded (NPAD rows/batch) layout
__global__ void k_normalize_x(const float* __restrict__ x,
                              float* __restrict__ xn,
                              _Float16* __restrict__ xnh) {
  __shared__ float red[256];
  int row = blockIdx.x, tid = threadIdx.x;     // row in [0, BN)
  int b = row / Nn, n = row % Nn;
  const float* xr = x + (size_t)row * Dm;
  float s2 = 0.f;
  for (int i = tid; i < Dm; i += 256) { float v = xr[i]; s2 += v * v; }
  red[tid] = s2; __syncthreads();
  for (int k = 128; k > 0; k >>= 1) { if (tid < k) red[tid] += red[tid + k]; __syncthreads(); }
  float inv = 1.f / fmaxf(sqrtf(red[0]), EPSN);
  _Float16* hr = xnh + ((size_t)b * NPAD + n) * Dm;
  for (int i = tid; i < Dm; i += 256) {
    float v = xr[i] * inv;
    xn[(size_t)row * Dm + i] = v;
    hr[i] = (_Float16)v;
  }
}

__global__ void k_normalize_c(const float* __restrict__ c,
                              float* __restrict__ cn) {
  __shared__ float red[256];
  int row = blockIdx.x, tid = threadIdx.x;
  const float* xr = c + (size_t)row * Dm;
  float s2 = 0.f;
  for (int i = tid; i < Dm; i += 256) { float v = xr[i]; s2 += v * v; }
  red[tid] = s2; __syncthreads();
  for (int k = 128; k > 0; k >>= 1) { if (tid < k) red[tid] += red[tid + k]; __syncthreads(); }
  float inv = 1.f / fmaxf(sqrtf(red[0]), EPSN);
  for (int i = tid; i < Dm; i += 256) cn[(size_t)row * Dm + i] = xr[i] * inv;
}

// ============== attention kernels ==============

// cross-attn: one block per (b, h, qi); 784 keys, dh=64
__global__ void k_cross_attn(const float* __restrict__ qq,   // (B,Kq,Dm)
                             const float* __restrict__ kk,   // (B,Nn,Dm)
                             const float* __restrict__ vv,
                             float* __restrict__ o) {        // (B,Kq,Dm)
  __shared__ float qrow[DH];
  __shared__ float sc[Nn];
  __shared__ float red[256];
  __shared__ float opart[4][DH];
  int bi = blockIdx.x;
  int b  = bi / (Hh * Kq);
  int rm = bi % (Hh * Kq);
  int h  = rm / Kq, qi = rm % Kq;
  int tid = threadIdx.x;
  const float scale = rsqrtf((float)Dm);   // scale by sqrt(d_model), per reference
  if (tid < DH) qrow[tid] = qq[((size_t)b * Kq + qi) * Dm + h * DH + tid];
  __syncthreads();
  float lm = -1e30f;
  for (int j = tid; j < Nn; j += 256) {
    const float* kr = kk + ((size_t)b * Nn + j) * Dm + h * DH;
    float s = 0.f;
#pragma unroll 8
    for (int d = 0; d < DH; ++d) s += qrow[d] * kr[d];
    s *= scale;
    sc[j] = s;
    lm = fmaxf(lm, s);
  }
  red[tid] = lm; __syncthreads();
  for (int k = 128; k > 0; k >>= 1) { if (tid < k) red[tid] = fmaxf(red[tid], red[tid + k]); __syncthreads(); }
  float mx = red[0]; __syncthreads();
  float ls = 0.f;
  for (int j = tid; j < Nn; j += 256) { float e = __expf(sc[j] - mx); sc[j] = e; ls += e; }
  red[tid] = ls; __syncthreads();
  for (int k = 128; k > 0; k >>= 1) { if (tid < k) red[tid] += red[tid + k]; __syncthreads(); }
  float inv = 1.f / red[0];
  __syncthreads();
  // all 256 threads accumulate: 4 j-chunks x 64 dims
  {
    int d = tid & 63, c = tid >> 6;
    float acc = 0.f;
    for (int j = c; j < Nn; j += 4)
      acc += sc[j] * vv[((size_t)b * Nn + j) * Dm + h * DH + d];
    opart[c][d] = acc;
  }
  __syncthreads();
  if (tid < DH) {
    float s = opart[0][tid] + opart[1][tid] + opart[2][tid] + opart[3][tid];
    o[((size_t)b * Kq + qi) * Dm + h * DH + tid] = s * inv;
  }
}

// self-attn on (B,Kq,Dm): one thread per (b,h,qi)
__global__ void k_self_attn(const float* __restrict__ qq,
                            const float* __restrict__ kk,
                            const float* __restrict__ vv,
                            float* __restrict__ o) {
  int t = blockIdx.x * blockDim.x + threadIdx.x;
  if (t >= Bb * Hh * Kq) return;
  int b = t / (Hh * Kq);
  int rm = t % (Hh * Kq);
  int h = rm / Kq, qi = rm % Kq;
  const float scale = rsqrtf((float)Dm);
  float s[Kq];
  float mx = -1e30f;
  for (int j = 0; j < Kq; ++j) {
    float acc = 0.f;
    const float* qr = qq + ((size_t)b * Kq + qi) * Dm + h * DH;
    const float* kr = kk + ((size_t)b * Kq + j) * Dm + h * DH;
    for (int d = 0; d < DH; ++d) acc += qr[d] * kr[d];
    s[j] = acc * scale;
    mx = fmaxf(mx, s[j]);
  }
  float sum = 0.f;
  for (int j = 0; j < Kq; ++j) { s[j] = __expf(s[j] - mx); sum += s[j]; }
  float inv = 1.f / sum;
  for (int d = 0; d < DH; ++d) {
    float acc = 0.f;
    for (int j = 0; j < Kq; ++j)
      acc += s[j] * vv[((size_t)b * Kq + j) * Dm + h * DH + d];
    o[((size_t)b * Kq + qi) * Dm + h * DH + d] = acc * inv;
  }
}

// ============== affinity / assignment finalization ==============

__global__ void k_S(const float* __restrict__ xn, const float* __restrict__ cn,
                    float* __restrict__ Sws, float* __restrict__ Sout) {
  int t = blockIdx.x * blockDim.x + threadIdx.x;
  if (t >= Bb * Nn * Kq) return;
  int b = t / (Nn * Kq);
  int rm = t % (Nn * Kq);
  int n = rm / Kq, k = rm % Kq;
  const float* xr = xn + ((size_t)b * Nn + n) * Dm;
  const float* cr = cn + ((size_t)b * Kq + k) * Dm;
  float acc = 0.f;
  for (int d = 0; d < Dm; ++d) acc += xr[d] * cr[d];
  acc = fmaxf(acc, 0.f);
  Sws[t] = acc;
  Sout[t] = acc;
}

__global__ void k_reduce_M(const float* __restrict__ Ksum,
                           float* __restrict__ Mbuf, float* __restrict__ Mout) {
  __shared__ float red[256];
  int b = blockIdx.x, tid = threadIdx.x;
  float s = 0.f;
  for (int i = tid; i < Nn; i += 256) s += Ksum[b * Nn + i];
  red[tid] = s; __syncthreads();
  for (int k = 128; k > 0; k >>= 1) { if (tid < k) red[tid] += red[tid + k]; __syncthreads(); }
  if (tid == 0) { Mbuf[b] = red[0]; Mout[b] = red[0]; }
}

__global__ void k_finalize_W(float* __restrict__ W,
                             const float* __restrict__ Ksum,
                             const float* __restrict__ Mbuf) {
  long long t = (long long)blockIdx.x * blockDim.x + threadIdx.x;
  if (t >= (long long)Bb * Nn * Nn) return;
  int b = (int)(t / ((long long)Nn * Nn));
  int rm = (int)(t % ((long long)Nn * Nn));
  int i = rm / Nn, j = rm % Nn;
  W[t] = W[t] - Ksum[b * Nn + i] * Ksum[b * Nn + j] / Mbuf[b];
}

__global__ void k_delta(const float* __restrict__ S, float* __restrict__ dlt) {
  long long t = (long long)blockIdx.x * blockDim.x + threadIdx.x;
  if (t >= (long long)Bb * Nn * Nn) return;
  int b = (int)(t / ((long long)Nn * Nn));
  int rm = (int)(t % ((long long)Nn * Nn));
  int i = rm / Nn, j = rm % Nn;
  const float* si = S + ((size_t)b * Nn + i) * Kq;
  const float* sj = S + ((size_t)b * Nn + j) * Kq;
  float m = -1e30f;
#pragma unroll
  for (int k = 0; k < Kq; ++k) m = fmaxf(m, si[k] * sj[k]);
  dlt[t] = m;
}

// ============== host launcher ==============

extern "C" void kernel_launch(void* const* d_in, const int* in_sizes, int n_in,
                              void* d_out, int out_size, void* d_ws, size_t ws_size,
                              hipStream_t stream) {
  const float* x      = (const float*)d_in[0];
  const float* protos = (const float*)d_in[1];
  // per-layer param indices (dict insertion order from setup_inputs)
  auto P = [&](int l, int k) -> const float* { return (const float*)d_in[2 + l * 26 + k]; };
  // cross: 0 wq.w 1 wq.b 2 wk.w 3 wk.b 4 wv.w 5 wv.b 6 wo.w 7 wo.b
  // self : 8..15 ; ffn: 16 l1.w 17 l1.b 18 l2.w 19 l2.b
  // ln1: 20 g 21 b ; ln2: 22 23 ; ln3: 24 25

  float* out = (float*)d_out;

  // ---- workspace bump allocator ----
  char* ws = (char*)d_ws;
  size_t off = 0;
  auto alloc = [&](size_t bytes) -> char* {
    char* p = ws + off;
    off = (off + bytes + 255) & ~(size_t)255;
    return p;
  };
  _Float16* xh = (_Float16*)alloc((size_t)BN * Dm * 2);
  _Float16* wkt[Lr]; _Float16* wvt[Lr];
  for (int l = 0; l < Lr; ++l) {
    wkt[l] = (_Float16*)alloc((size_t)Dm * Dm * 2);
    wvt[l] = (_Float16*)alloc((size_t)Dm * Dm * 2);
  }
  float* kk = (float*)alloc((size_t)BN * Dm * 4);   // 19.3 MB
  float* vv = (float*)alloc((size_t)BN * Dm * 4);   // 19.3 MB
  float* conc0 = (float*)alloc(Bb * Kq * Dm * 4);
  float* conc1 = (float*)alloc(Bb * Kq * Dm * 4);
  float* qqbuf = (float*)alloc(Bb * Kq * Dm * 4);
  float* obuf  = (float*)alloc(Bb * Kq * Dm * 4);
  float* tmp1  = (float*)alloc(Bb * Kq * Dm * 4);
  float* x1buf = (float*)alloc(Bb * Kq * Dm * 4);
  float* x2buf = (float*)alloc(Bb * Kq * Dm * 4);
  float* kks   = (float*)alloc(Bb * Kq * Dm * 4);
  float* vvs   = (float*)alloc(Bb * Kq * Dm * 4);
  float* hbuf  = (float*)alloc(Bb * Kq * DFf * 4);
  float* cnbuf = (float*)alloc(Bb * Kq * Dm * 4);
  float* Ksum  = (float*)alloc(Bb * Nn * 4);
  float* Mbuf  = (float*)alloc(Bb * 4);
  float* Sws   = (float*)alloc((size_t)Bb * Nn * Kq * 4);
  // phase-B aliases over kk/vv (dead after ACG layers)
  float*    xn  = kk;                         // 19.3 MB (fits kk region)
  _Float16* xnh = (_Float16*)vv;              // padded f16, 10.2 MB (fits vv region)

  const int T = 256;
  auto cdiv = [](long long a, long long b) { return (int)((a + b - 1) / b); };

  // ---- stage f16 operands ----
  k_f32_to_f16<<<cdiv((long long)BN * Dm, T), T, 0, stream>>>(x, xh, BN * Dm);
  for (int l = 0; l < Lr; ++l) {
    k_transpose_f16<<<cdiv(Dm * Dm, T), T, 0, stream>>>(P(l, 2), wkt[l], Dm, Dm);
    k_transpose_f16<<<cdiv(Dm * Dm, T), T, 0, stream>>>(P(l, 4), wvt[l], Dm, Dm);
  }
  k_broadcast_protos<<<cdiv(Bb * Kq * Dm, T), T, 0, stream>>>(protos, conc0);

  float* cur = conc0; float* nxt = conc1;
  const int MR = Bb * Kq;            // 80 concept rows
  const int gemm_blocks = cdiv((long long)(BN / 32) * (Dm / 64), 8);  // 8 waves/block

  for (int l = 0; l < Lr; ++l) {
    // --- cross attention (K/V over full x via blocked WMMA) ---
    k_wmma_gemm_nt24<<<gemm_blocks, T, 0, stream>>>(xh, wkt[l], P(l, 3), kk, BN, Dm, Dm);
    k_wmma_gemm_nt24<<<gemm_blocks, T, 0, stream>>>(xh, wvt[l], P(l, 5), vv, BN, Dm, Dm);
    k_small_gemm<<<cdiv(MR * Dm, T), T, 0, stream>>>(cur, P(l, 0), P(l, 1), nullptr, qqbuf, MR, Dm, Dm, 0);
    k_cross_attn<<<Bb * Hh * Kq, T, 0, stream>>>(qqbuf, kk, vv, obuf);
    k_small_gemm<<<cdiv(MR * Dm, T), T, 0, stream>>>(obuf, P(l, 6), P(l, 7), qqbuf, tmp1, MR, Dm, Dm, 0);
    k_ln_resid<<<MR, T, 0, stream>>>(tmp1, P(l, 20), P(l, 21), cur, x1buf);
    // --- self attention on concepts ---
    k_small_gemm<<<cdiv(MR * Dm, T), T, 0, stream>>>(x1buf, P(l, 8),  P(l, 9),  nullptr, qqbuf, MR, Dm, Dm, 0);
    k_small_gemm<<<cdiv(MR * Dm, T), T, 0, stream>>>(x1buf, P(l, 10), P(l, 11), nullptr, kks,   MR, Dm, Dm, 0);
    k_small_gemm<<<cdiv(MR * Dm, T), T, 0, stream>>>(x1buf, P(l, 12), P(l, 13), nullptr, vvs,   MR, Dm, Dm, 0);
    k_self_attn<<<cdiv(Bb * Hh * Kq, T), T, 0, stream>>>(qqbuf, kks, vvs, obuf);
    k_small_gemm<<<cdiv(MR * Dm, T), T, 0, stream>>>(obuf, P(l, 14), P(l, 15), qqbuf, tmp1, MR, Dm, Dm, 0);
    k_ln_resid<<<MR, T, 0, stream>>>(tmp1, P(l, 22), P(l, 23), x1buf, x2buf);
    // --- FFN ---
    k_small_gemm<<<cdiv(MR * DFf, T), T, 0, stream>>>(x2buf, P(l, 16), P(l, 17), nullptr, hbuf, MR, Dm, DFf, 1);
    k_small_gemm<<<cdiv(MR * Dm, T), T, 0, stream>>>(hbuf, P(l, 18), P(l, 19), nullptr, tmp1, MR, DFf, Dm, 0);
    k_ln_resid<<<MR, T, 0, stream>>>(tmp1, P(l, 24), P(l, 25), x2buf, nxt);
    float* t = cur; cur = nxt; nxt = t;
  }

  // ---- outputs: concepts ----
  k_copy_f32<<<cdiv(Bb * Kq * Dm, T), T, 0, stream>>>(cur, out + C_OFF, Bb * Kq * Dm);

  // ---- phase B: affinity graph + pixel assignment ----
  k_zero_f16<<<cdiv((long long)Bb * NPAD * Dm, T), T, 0, stream>>>(xnh, Bb * NPAD * Dm);
  k_normalize_x<<<BN, T, 0, stream>>>(x, xn, xnh);
  k_normalize_c<<<MR, T, 0, stream>>>(cur, cnbuf);
  k_S<<<cdiv(Bb * Nn * Kq, T), T, 0, stream>>>(xn, cnbuf, Sws, out + S_OFF);
  k_zero_f32<<<cdiv(Bb * Nn, T), T, 0, stream>>>(Ksum, Bb * Nn);
  {
    dim3 g(cdiv((NPAD / 32) * (NPAD / 64), 8), Bb);
    k_wmma_gram24<<<g, T, 0, stream>>>(xnh, out + W_OFF, Ksum);
  }
  k_reduce_M<<<Bb, T, 0, stream>>>(Ksum, Mbuf, out + M_OFF);
  k_finalize_W<<<cdiv((long long)Bb * Nn * Nn, T), T, 0, stream>>>(out + W_OFF, Ksum, Mbuf);
  k_delta<<<cdiv((long long)Bb * Nn * Nn, T), T, 0, stream>>>(Sws, out + DLT_OFF);
}
